// DiffFormer_63153199121059
// MI455X (gfx1250) — compile-verified, matrix operand order checked by reading, not run on hardware
//
#include <hip/hip_runtime.h>

typedef __attribute__((ext_vector_type(16))) _Float16 v16h;
typedef __attribute__((ext_vector_type(8)))  _Float16 v8h;
typedef __attribute__((ext_vector_type(8)))  float    v8f;

constexpr int D_MODEL  = 128;
constexpr int D_STATE  = 16;
constexpr int D_INNER  = 256;
constexpr int DT_RANK  = 8;
constexpr int BATCH    = 128;
constexpr int SEQ      = 200;
constexpr int BL       = BATCH * SEQ;     // 25600 rows (multiple of 16)
constexpr int N_LAYERS = 2;
constexpr int GRID     = 16;
constexpr int KKAN     = 2 * D_MODEL * GRID; // 4096 (cos||sin features)
constexpr int NXP      = 48;              // xproj output padded 40 -> 48

#define DEV __device__ __forceinline__

union FragH { v16h v; v8h h[2]; };

DEV v8f wmma16(v16h a, v16h b, v8f c) {
    // D = A(16x32 f16) x B(32x16 f16) + C(16x16 f32)
    return __builtin_amdgcn_wmma_f32_16x16x32_f16(
        /*neg_a=*/false, a, /*neg_b=*/false, b,
        /*c_mod=*/(short)0, c, /*reuse_a=*/false, /*reuse_b=*/false);
}

// ---------------------------------------------------------------- utilities
__global__ void k_zero(float* __restrict__ o, int n) {
    int i = blockIdx.x * 256 + threadIdx.x;
    if (i < n) o[i] = 0.0f;
}

__global__ void k_cvt(const float* __restrict__ a, _Float16* __restrict__ o, int n) {
    int i = blockIdx.x * 256 + threadIdx.x;
    if (i < n) o[i] = (_Float16)a[i];
}

// xproj weight (40,256) -> padded f16 (48,256), rows 40..47 zero
__global__ void k_pack_xproj(const float* __restrict__ w, _Float16* __restrict__ o) {
    int i = blockIdx.x * 256 + threadIdx.x;
    if (i >= NXP * D_INNER) return;
    int r = i >> 8; // row / 256
    o[i] = (_Float16)(r < (DT_RANK + 2 * D_STATE) ? w[i] : 0.0f);
}

// kan_coef (2,128,128,16) f32 -> combined (128, 4096) f16:
// out[j][i*16+g] = coef[0][j][i][g];  out[j][2048 + i*16+g] = coef[1][j][i][g]
__global__ void k_pack_kan(const float* __restrict__ c, _Float16* __restrict__ o) {
    int i = blockIdx.x * 256 + threadIdx.x;
    if (i >= D_MODEL * KKAN) return;
    int j = i >> 12, col = i & (KKAN - 1);
    int part = col >> 11, rem = col & 2047;
    int ii = rem >> 4, g = rem & 15;
    o[i] = (_Float16)c[(((part * D_MODEL + j) * D_MODEL) + ii) * GRID + g];
}

// ---------------------------------------------------------------- LayerNorm
// one row (128 wide) per 128-thread block; optional f32 and f16 outputs
__global__ void k_ln(const float* __restrict__ X, const float* __restrict__ w,
                     const float* __restrict__ b, float* __restrict__ Yf,
                     _Float16* __restrict__ Yh) {
    __shared__ float red[D_MODEL];
    int row = blockIdx.x, t = threadIdx.x;
    float v = X[(size_t)row * D_MODEL + t];
    red[t] = v; __syncthreads();
    for (int s = 64; s > 0; s >>= 1) { if (t < s) red[t] += red[t + s]; __syncthreads(); }
    float mu = red[0] * (1.0f / D_MODEL);
    __syncthreads();
    float d = v - mu;
    red[t] = d * d; __syncthreads();
    for (int s = 64; s > 0; s >>= 1) { if (t < s) red[t] += red[t + s]; __syncthreads(); }
    float var = red[0] * (1.0f / D_MODEL);
    float y = w[t] * d * rsqrtf(var + 1e-12f) + b[t];
    if (Yf) Yf[(size_t)row * D_MODEL + t] = y;
    if (Yh) Yh[(size_t)row * D_MODEL + t] = (_Float16)y;
}

// ---------------------------------------------------------------- WMMA GEMM
// C(M,Nstride) f32 = A(M,K) f16  x  B(N,K) f16 ^T
// grid.x = M/16 ; 4 waves/block each owning one 16-wide N tile
__global__ void k_gemm(const _Float16* __restrict__ A, const _Float16* __restrict__ B,
                       float* __restrict__ C, int K, int Nstride, int nTilesTotal) {
    int wave  = threadIdx.x >> 5;
    int nTile = blockIdx.y * 4 + wave;
    if (nTile >= nTilesTotal) return;
    int lane = threadIdx.x & 31;
    int idx  = lane & 15;      // A: M row index / B: N row index / D: N column
    int half = lane >> 4;

    const _Float16* arow = A + ((size_t)(blockIdx.x * 16 + idx)) * K;
    const _Float16* brow = B + ((size_t)(nTile * 16 + idx)) * K;
    __builtin_prefetch(brow, 0, 1);   // global_prefetch_b8 on weight row

    v8f c = {};
    for (int k = 0; k < K; k += 32) {
        FragH a, bfrag;
        a.h[0]     = *(const v8h*)(arow + k + half * 8);
        a.h[1]     = *(const v8h*)(arow + k + 16 + half * 8);
        bfrag.h[0] = *(const v8h*)(brow + k + half * 8);
        bfrag.h[1] = *(const v8h*)(brow + k + 16 + half * 8);
        c = wmma16(a.v, bfrag.v, c);
    }
    int mbase = blockIdx.x * 16 + (half ? 8 : 0);
    float* crow = C + (size_t)0;
    for (int v = 0; v < 8; ++v)
        crow[(size_t)(mbase + v) * Nstride + nTile * 16 + idx] = c[v];
}

// ---------------------------------------------------------------- conv+SiLU
// xc = silu(depthwise-causal-conv4(xi) + bias); rev=1 runs on time-reversed xi
// output stored in scan-time order; one thread per (b,t,d)
__global__ void k_conv(const float* __restrict__ XZ, const float* __restrict__ cw,
                       const float* __restrict__ cb, _Float16* __restrict__ XC, int rev) {
    int d  = threadIdx.x;
    int bt = blockIdx.x;
    int b  = bt / SEQ, t = bt - b * SEQ;
    float acc = cb[d];
    for (int k = 0; k < 4; ++k) {
        int p = t - 3 + k;
        if (p >= 0) {
            int tn = rev ? (SEQ - 1 - p) : p;
            acc += XZ[((size_t)(b * SEQ + tn)) * (2 * D_INNER) + d] * cw[d * 4 + k];
        }
    }
    float s = acc / (1.0f + __expf(-acc));
    XC[(size_t)bt * D_INNER + d] = (_Float16)s;
}

// ---------------------------------------------------------------- scan
// one block per batch, one thread per channel d; fused dtproj+softplus+gating.
// fwd (rev=0): G = val ; bwd (rev=1,fin=1): G += val and emit f16 copy.
__global__ void k_scan(const float* __restrict__ dbl, const _Float16* __restrict__ xc,
                       const float* __restrict__ xz, const float* __restrict__ dtw,
                       const float* __restrict__ dtb, const float* __restrict__ Alog,
                       const float* __restrict__ Dp, float* __restrict__ G,
                       _Float16* __restrict__ Gh, int rev, int fin) {
    __shared__ float sm[NXP];
    int b = blockIdx.x, d = threadIdx.x;
    float A[D_STATE], h[D_STATE], w[DT_RANK];
    for (int s = 0; s < D_STATE; ++s) { A[s] = -__expf(Alog[d * D_STATE + s]); h[s] = 0.0f; }
    for (int r = 0; r < DT_RANK; ++r) w[r] = dtw[d * DT_RANK + r];
    float bias = dtb[d], Dd = Dp[d];

    for (int t = 0; t < SEQ; ++t) {
        if (threadIdx.x < NXP)
            sm[threadIdx.x] = dbl[((size_t)(b * SEQ + t)) * NXP + threadIdx.x];
        __syncthreads();
        float dta = bias;
        for (int r = 0; r < DT_RANK; ++r) dta += sm[r] * w[r];
        float dt = (dta > 20.0f) ? dta : log1pf(__expf(dta));     // softplus
        float u = (float)xc[((size_t)(b * SEQ + t)) * D_INNER + d];
        float y = 0.0f;
        for (int s = 0; s < D_STATE; ++s) {
            float hs = __expf(dt * A[s]) * h[s] + (dt * u) * sm[DT_RANK + s];
            h[s] = hs;
            y += hs * sm[DT_RANK + D_STATE + s];
        }
        int tn = rev ? (SEQ - 1 - t) : t;                          // natural time index
        float z  = xz[((size_t)(b * SEQ + tn)) * (2 * D_INNER) + D_INNER + d];
        float sz = z / (1.0f + __expf(-z));
        float val = (y + u * Dd) * sz;
        size_t gi = ((size_t)(b * SEQ + tn)) * D_INNER + d;
        float g = rev ? (G[gi] + val) : val;
        G[gi] = g;
        if (fin) Gh[gi] = (_Float16)g;
        __syncthreads();
    }
}

// ---------------------------------------------------------------- KAN + residual
// Block = 16 output rows x full 128 N; 8 waves each own a 16-wide N tile.
// Phase 0: cos features (16x2048 f16) in LDS, phase 1: sin features; each
// phase consumed by a 64-iteration WMMA K-loop against coef half.
__global__ void __launch_bounds__(256)
k_kan(const float* __restrict__ O2, const _Float16* __restrict__ Wk,
      const float* __restrict__ hin, float* __restrict__ hout,
      float* __restrict__ out) {
    extern __shared__ _Float16 sfeat[];   // 16 * 2048 f16 = 64 KB
    int m0   = blockIdx.x * 16;
    int wave = threadIdx.x >> 5;          // nTile 0..7 -> N = 128
    int lane = threadIdx.x & 31;
    int idx  = lane & 15;
    int half = lane >> 4;

    v8f c = {};
    for (int ph = 0; ph < 2; ++ph) {
        __syncthreads();
        // fill 16x2048 features: 2048 (row,i) pairs over 256 threads
        for (int j = 0; j < 8; ++j) {
            int p = threadIdx.x + 256 * j;
            int r = p >> 7, i = p & 127;
            float xv = O2[(size_t)(m0 + r) * D_MODEL + i];
            _Float16* dst = sfeat + r * 2048 + i * GRID;
            for (int g = 0; g < GRID; ++g) {
                float ang = xv * (float)(g + 1);
                dst[g] = (_Float16)(ph == 0 ? __cosf(ang) : __sinf(ang));
            }
        }
        __syncthreads();
        const _Float16* lrow = sfeat + idx * 2048;                    // A from LDS
        const _Float16* brow = Wk + (size_t)(wave * 16 + idx) * KKAN + ph * 2048;
        for (int k = 0; k < 2048; k += 32) {
            FragH a, bfrag;
            a.h[0]     = *(const v8h*)(lrow + k + half * 8);          // ds_load_b128
            a.h[1]     = *(const v8h*)(lrow + k + 16 + half * 8);
            bfrag.h[0] = *(const v8h*)(brow + k + half * 8);
            bfrag.h[1] = *(const v8h*)(brow + k + 16 + half * 8);
            c = wmma16(a.v, bfrag.v, c);
        }
    }
    int mbase = m0 + (half ? 8 : 0);
    int col   = wave * 16 + idx;
    for (int v = 0; v < 8; ++v) {
        size_t o  = (size_t)(mbase + v) * D_MODEL + col;
        float hn = c[v] + hin[o];           // residual
        hout[o] = hn;
        out[o] += hn * (1.0f / N_LAYERS);   // mean over layer features
    }
}

// ---------------------------------------------------------------- host
extern "C" void kernel_launch(void* const* d_in, const int* in_sizes, int n_in,
                              void* d_out, int out_size, void* d_ws, size_t ws_size,
                              hipStream_t stream) {
    const float* x     = (const float*)d_in[0];
    const float* in_w  = (const float*)d_in[1];
    const float* out_w = (const float*)d_in[2];
    const float* cw_f  = (const float*)d_in[3];
    const float* cb_f  = (const float*)d_in[4];
    const float* cw_b  = (const float*)d_in[5];
    const float* cb_b  = (const float*)d_in[6];
    const float* xpw_f = (const float*)d_in[7];
    const float* xpw_b = (const float*)d_in[8];
    const float* dtw_f = (const float*)d_in[9];
    const float* dtb_f = (const float*)d_in[10];
    const float* dtw_b = (const float*)d_in[11];
    const float* dtb_b = (const float*)d_in[12];
    const float* Al_f  = (const float*)d_in[13];
    const float* Al_b  = (const float*)d_in[14];
    const float* D_f   = (const float*)d_in[15];
    const float* D_b   = (const float*)d_in[16];
    const float* ln1w  = (const float*)d_in[17];
    const float* ln1b  = (const float*)d_in[18];
    const float* lnw   = (const float*)d_in[19];
    const float* lnb   = (const float*)d_in[20];
    const float* kanc  = (const float*)d_in[21];

    char* wsp = (char*)d_ws;
    auto alloc = [&](size_t bytes) -> char* {
        char* p = wsp; wsp += (bytes + 255) & ~(size_t)255; return p;
    };
    float*    H    = (float*)   alloc((size_t)BL * D_MODEL * 4);   // residual stream
    _Float16* Oh   = (_Float16*)alloc((size_t)BL * D_MODEL * 2);   // ln1 out (f16)
    float*    XZ   = (float*)   alloc((size_t)BL * 2 * D_INNER * 4);
    _Float16* XCf  = (_Float16*)alloc((size_t)BL * D_INNER * 2);
    _Float16* XCb  = (_Float16*)alloc((size_t)BL * D_INNER * 2);
    float*    DBf  = (float*)   alloc((size_t)BL * NXP * 4);
    float*    DBb  = (float*)   alloc((size_t)BL * NXP * 4);
    float*    G    = (float*)   alloc((size_t)BL * D_INNER * 4);
    _Float16* Gh   = (_Float16*)alloc((size_t)BL * D_INNER * 2);
    float*    MID  = (float*)   alloc((size_t)BL * D_MODEL * 4);
    float*    O2   = (float*)   alloc((size_t)BL * D_MODEL * 4);
    _Float16* inw16  = (_Float16*)alloc((size_t)2 * D_INNER * D_MODEL * 2);
    _Float16* outw16 = (_Float16*)alloc((size_t)D_MODEL * D_INNER * 2);
    _Float16* xpf16  = (_Float16*)alloc((size_t)NXP * D_INNER * 2);
    _Float16* xpb16  = (_Float16*)alloc((size_t)NXP * D_INNER * 2);
    _Float16* kanw16 = (_Float16*)alloc((size_t)D_MODEL * KKAN * 2);

    const int MT = BL / 16;  // 1600 M tiles
    k_zero<<<(BL * D_MODEL + 255) / 256, 256, 0, stream>>>((float*)d_out, BL * D_MODEL);

    for (int l = 0; l < N_LAYERS; ++l) {
        // weight prep (f16 packing)
        k_cvt<<<(2 * D_INNER * D_MODEL + 255) / 256, 256, 0, stream>>>(
            in_w + (size_t)l * 2 * D_INNER * D_MODEL, inw16, 2 * D_INNER * D_MODEL);
        k_cvt<<<(D_MODEL * D_INNER + 255) / 256, 256, 0, stream>>>(
            out_w + (size_t)l * D_MODEL * D_INNER, outw16, D_MODEL * D_INNER);
        k_pack_xproj<<<(NXP * D_INNER + 255) / 256, 256, 0, stream>>>(
            xpw_f + (size_t)l * (DT_RANK + 2 * D_STATE) * D_INNER, xpf16);
        k_pack_xproj<<<(NXP * D_INNER + 255) / 256, 256, 0, stream>>>(
            xpw_b + (size_t)l * (DT_RANK + 2 * D_STATE) * D_INNER, xpb16);
        k_pack_kan<<<(D_MODEL * KKAN + 255) / 256, 256, 0, stream>>>(
            kanc + (size_t)l * 2 * D_MODEL * D_MODEL * GRID, kanw16);

        const float* hin = (l == 0) ? x : H;

        // ln1 -> f16 activations
        k_ln<<<BL, D_MODEL, 0, stream>>>(hin, ln1w + l * D_MODEL, ln1b + l * D_MODEL,
                                         nullptr, Oh);
        // in_proj: (BL,128) x (512,128)^T -> XZ (BL,512); shared by fwd/bwd
        k_gemm<<<dim3(MT, 8), 128, 0, stream>>>(Oh, inw16, XZ, D_MODEL, 2 * D_INNER, 32);
        // depthwise conv + silu (fwd natural order, bwd reversed order)
        k_conv<<<BL, D_INNER, 0, stream>>>(XZ, cw_f + l * D_INNER * 4, cb_f + l * D_INNER, XCf, 0);
        k_conv<<<BL, D_INNER, 0, stream>>>(XZ, cw_b + l * D_INNER * 4, cb_b + l * D_INNER, XCb, 1);
        // xproj: (BL,256) x (48,256)^T -> dbl (BL,48)
        k_gemm<<<dim3(MT, 1), 128, 0, stream>>>(XCf, xpf16, DBf, D_INNER, NXP, 3);
        k_gemm<<<dim3(MT, 1), 128, 0, stream>>>(XCb, xpb16, DBb, D_INNER, NXP, 3);
        // selective scans (fused dtproj + softplus + gating); bwd accumulates
        k_scan<<<BATCH, D_INNER, 0, stream>>>(DBf, XCf, XZ,
            dtw_f + l * D_INNER * DT_RANK, dtb_f + l * D_INNER,
            Al_f + l * D_INNER * D_STATE, D_f + l * D_INNER, G, Gh, 0, 0);
        k_scan<<<BATCH, D_INNER, 0, stream>>>(DBb, XCb, XZ,
            dtw_b + l * D_INNER * DT_RANK, dtb_b + l * D_INNER,
            Al_b + l * D_INNER * D_STATE, D_b + l * D_INNER, G, Gh, 1, 1);
        // out_proj: (BL,256) x (128,256)^T -> MID (BL,128)
        k_gemm<<<dim3(MT, 2), 128, 0, stream>>>(Gh, outw16, MID, D_INNER, D_MODEL, 8);
        // ln2
        k_ln<<<BL, D_MODEL, 0, stream>>>(MID, lnw + l * D_MODEL, lnb + l * D_MODEL, O2, nullptr);
        // KAN (LDS feature staging + WMMA) + residual + feature-mean accumulation
        k_kan<<<MT, 256, 16 * 2048 * sizeof(_Float16), stream>>>(O2, kanw16, hin, H,
                                                                 (float*)d_out);
    }
}
